// DCGRUClassifier_84576495993173
// MI455X (gfx1250) — compile-verified, matrix operand order checked by reading, not run on hardware
//
#include <hip/hip_runtime.h>
#include <hip/hip_bf16.h>
#include <math.h>

// ---------------- problem constants ----------------
#define NB   32            // batch
#define TT   128           // timesteps
#define NN   256           // graph nodes
#define FIN  64            // input dim (== UNITS, also layer-1 input dim)
#define UU   64            // GRU units
#define MM   3             // Chebyshev matrices (K+1)
#define FF   128           // concat feature dim (FIN + UU)
#define CC   (FF * NB)     // 4096: columns of the diffusion matmul (N, F*B)
#define ROWS (NB * NN)     // 8192: rows of the projection matmul
#define KG   (FF * MM)     // 384: reduction dim of the projection matmul
#define NCLS 4

typedef __bf16 bf16_t;
typedef __attribute__((ext_vector_type(16))) __bf16 v16bf;
typedef __attribute__((ext_vector_type(8)))  __bf16 v8bf;
typedef __attribute__((ext_vector_type(8)))  float  v8f;

static __device__ __forceinline__ bf16_t f2bf(float f) { return static_cast<bf16_t>(f); }
static __device__ __forceinline__ float  bf2f(bf16_t v) { return static_cast<float>(v); }

// Fragment loader for BOTH operands. Source is K-contiguous with leading dim ld
// (A: row-major [M x K]; B: stored transposed as [N x K]).
// ISA 7.12.2 wave32 16-bit layout: per lane the 16 bf16 values are the two
// contiguous 8-element chunks at K = k0 + 8*hi and K = k0 + 16 + 8*hi.
// Both chunks are 16-byte aligned (ld and k0 are multiples of 8 elements),
// so each compiles to one global_load_b128.
static __device__ __forceinline__ v16bf load_frag(const bf16_t* __restrict__ base,
                                                  int ld, int rowcol, int k0, int hi) {
  const bf16_t* p = base + (size_t)rowcol * ld + k0 + 8 * hi;
  v8bf x = *(const v8bf*)(p);
  v8bf y = *(const v8bf*)(p + 16);
  return __builtin_shufflevector(x, y, 0, 1, 2, 3, 4, 5, 6, 7,
                                 8, 9, 10, 11, 12, 13, 14, 15);
}

// ---------------- f32 -> bf16 conversion ----------------
__global__ void f32_to_bf16(bf16_t* __restrict__ dst, const float* __restrict__ src, int n) {
  int i = blockIdx.x * blockDim.x + threadIdx.x;
  if (i < n) dst[i] = f2bf(src[i]);
}

// Transposing convert for weights: dst (outn x KG, K-contiguous) <- src (KG x outn)
__global__ void f32_to_bf16_T(bf16_t* __restrict__ dst, const float* __restrict__ src,
                              int outn) {
  int idx = blockIdx.x * blockDim.x + threadIdx.x;
  if (idx >= KG * outn) return;
  int o = idx / KG, k = idx % KG;
  dst[idx] = f2bf(src[k * outn + o]);
}

// ---------------- Chebyshev diffusion: Yt = bf16(ALPHA*(S@X) - (SUB?Z:0)) ----------------
// S: (NN x NN) bf16 row-major (A operand). Xt,Zt,Yt: (CC x NN) bf16, i.e. the
// diffusion matrix stored column-major so B fragments are K-contiguous.
// 128 threads = 4 waves stacked along M; each wave owns 4 accumulator tiles
// (16 rows x 64 cols), so one A fragment feeds 4 WMMAs.
template <int ALPHA, bool SUB>
__global__ __launch_bounds__(128) void cheb_wmma(const bf16_t* __restrict__ S,
                                                 const bf16_t* __restrict__ Xt,
                                                 const bf16_t* __restrict__ Zt,
                                                 bf16_t* __restrict__ Yt) {
  const int wave = threadIdx.x >> 5, lane = threadIdx.x & 31;
  const int lo = lane & 15, hi = lane >> 4;
  const int m0 = (blockIdx.y * 4 + wave) * 16;
  const int n0 = blockIdx.x * 64;
  v8f acc[4] = {{}, {}, {}, {}};
#pragma unroll 2
  for (int k0 = 0; k0 < NN; k0 += 32) {
    v16bf a = load_frag(S, NN, m0 + lo, k0, hi);
#pragma unroll
    for (int j = 0; j < 4; ++j) {
      v16bf b = load_frag(Xt, NN, n0 + 16 * j + lo, k0, hi);
      acc[j] = __builtin_amdgcn_wmma_f32_16x16x32_bf16(false, a, false, b, (short)0,
                                                       acc[j], false, false);
    }
  }
#pragma unroll
  for (int j = 0; j < 4; ++j) {
#pragma unroll
    for (int i = 0; i < 8; ++i) {
      int row = m0 + i + 8 * hi;          // node index
      int col = n0 + 16 * j + lo;         // (f*NB + b) index
      float v = (float)ALPHA * acc[j][i];
      if (SUB) v -= bf2f(Zt[(size_t)col * NN + row]);
      Yt[(size_t)col * NN + row] = f2bf(v);
    }
  }
}

// ---------------- projection GEMM with fused GRU epilogue ----------------
// A: (ROWS x KG) bf16 row-major. Wt: (outn x KG) bf16 (K-contiguous). bias: f32.
// MODE 0: gates[row*128 + col] = sigmoid(acc + bias)           (outn = 128)
// MODE 1: state[row*64 + col]  = u*state + (1-u)*tanh(acc+b)   (outn = 64,
//          u = gates[row*128 + 64 + col])
template <int MODE>
__global__ __launch_bounds__(128) void proj_wmma(const bf16_t* __restrict__ A,
                                                 const bf16_t* __restrict__ Wt,
                                                 const float* __restrict__ bias,
                                                 float* __restrict__ gates,
                                                 float* __restrict__ state) {
  const int wave = threadIdx.x >> 5, lane = threadIdx.x & 31;
  const int lo = lane & 15, hi = lane >> 4;
  const int m0 = (blockIdx.y * 4 + wave) * 16;
  const int n0 = blockIdx.x * 64;
  v8f acc[4] = {{}, {}, {}, {}};
#pragma unroll 2
  for (int k0 = 0; k0 < KG; k0 += 32) {
    v16bf a = load_frag(A, KG, m0 + lo, k0, hi);
#pragma unroll
    for (int j = 0; j < 4; ++j) {
      v16bf b = load_frag(Wt, KG, n0 + 16 * j + lo, k0, hi);
      acc[j] = __builtin_amdgcn_wmma_f32_16x16x32_bf16(false, a, false, b, (short)0,
                                                       acc[j], false, false);
    }
  }
#pragma unroll
  for (int j = 0; j < 4; ++j) {
    const int col = n0 + 16 * j + lo;
    const float bv = bias[col];
#pragma unroll
    for (int i = 0; i < 8; ++i) {
      int row = m0 + i + 8 * hi;
      float v = acc[j][i] + bv;
      if (MODE == 0) {
        gates[row * (2 * UU) + col] = 1.0f / (1.0f + __expf(-v));
      } else {
        float u = gates[row * (2 * UU) + UU + col];
        float s = state[row * UU + col];
        state[row * UU + col] = u * s + (1.0f - u) * tanhf(v);
      }
    }
  }
}

// -------- pack x_cat into transposed diffusion layout: X0t[c*NN + n] --------
// c = f*NB + b ; value = f < FIN ? inp[b,n,f] : state[b,n,f-FIN] (optionally * r)
__global__ void pack_x0(bf16_t* __restrict__ X0t, const float* __restrict__ p0,
                        long p0_bstride, const float* __restrict__ state,
                        const float* __restrict__ gates /* nullptr => no r gate */) {
  int idx = blockIdx.x * blockDim.x + threadIdx.x;
  if (idx >= NN * CC) return;
  int n = idx % NN, c = idx / NN;      // write is contiguous in n
  int f = c / NB, b = c % NB;
  float v;
  if (f < FIN) {
    v = p0[(long)b * p0_bstride + n * FIN + f];
  } else {
    int fs = f - FIN;
    v = state[((long)b * NN + n) * UU + fs];
    if (gates) v *= gates[((long)b * NN + n) * (2 * UU) + fs];  // r gate (first half)
  }
  X0t[idx] = f2bf(v);
}

// -------- gather Chebyshev stack into (B*N, F*M) row-major --------
// Xg[b*NN+n, f*MM+m] = Xm_t[(f*NB + b)*NN + n]
__global__ void pack_xg(bf16_t* __restrict__ Xg, const bf16_t* __restrict__ X0t,
                        const bf16_t* __restrict__ X1t, const bf16_t* __restrict__ X2t) {
  int idx = blockIdx.x * blockDim.x + threadIdx.x;
  if (idx >= ROWS * KG) return;
  int row = idx / KG, q = idx % KG;
  int f = q / MM, m = q % MM;
  int b = row / NN, n = row % NN;
  const bf16_t* Xm = (m == 0) ? X0t : ((m == 1) ? X1t : X2t);
  Xg[idx] = Xm[(size_t)(f * NB + b) * NN + n];
}

// ---------------- FC head + max over nodes ----------------
__global__ __launch_bounds__(256) void head_max(const float* __restrict__ state1,
                                                const float* __restrict__ Wfc,
                                                const float* __restrict__ bfc,
                                                float* __restrict__ out) {
  __shared__ float red[256][NCLS];
  const int b = blockIdx.x, n = threadIdx.x;
  float acc[NCLS];
#pragma unroll
  for (int c = 0; c < NCLS; ++c) acc[c] = bfc[c];
  const float* s = state1 + ((long)b * NN + n) * UU;
  for (int f = 0; f < UU; ++f) {
    float v = fmaxf(s[f], 0.0f);
#pragma unroll
    for (int c = 0; c < NCLS; ++c) acc[c] += v * Wfc[f * NCLS + c];
  }
#pragma unroll
  for (int c = 0; c < NCLS; ++c) red[n][c] = acc[c];
  __syncthreads();
  for (int st = 128; st > 0; st >>= 1) {
    if (n < st)
#pragma unroll
      for (int c = 0; c < NCLS; ++c) red[n][c] = fmaxf(red[n][c], red[n + st][c]);
    __syncthreads();
  }
  if (n < NCLS) out[b * NCLS + n] = red[0][n];
}

extern "C" void kernel_launch(void* const* d_in, const int* in_sizes, int n_in,
                              void* d_out, int out_size, void* d_ws, size_t ws_size,
                              hipStream_t stream) {
  (void)in_sizes; (void)n_in; (void)out_size; (void)ws_size;
  const float* input_seq = (const float*)d_in[0];
  /* d_in[1] seq_lengths: unused by the reference forward */
  const float* supports  = (const float*)d_in[2];
  const float* Wg0 = (const float*)d_in[3];
  const float* bg0 = (const float*)d_in[4];
  const float* Wc0 = (const float*)d_in[5];
  const float* bc0 = (const float*)d_in[6];
  const float* Wg1 = (const float*)d_in[7];
  const float* bg1 = (const float*)d_in[8];
  const float* Wc1 = (const float*)d_in[9];
  const float* bc1 = (const float*)d_in[10];
  const float* Wfc = (const float*)d_in[11];
  const float* bfc = (const float*)d_in[12];

  // ---- workspace carve-out (≈21.4 MB total) ----
  char* ws = (char*)d_ws;
  size_t cur = 0;
  auto alloc = [&](size_t bytes) -> void* {
    void* p = ws + cur;
    cur = (cur + bytes + 255) & ~(size_t)255;
    return p;
  };
  float*  state0 = (float*)alloc((size_t)ROWS * UU * 4);
  float*  state1 = (float*)alloc((size_t)ROWS * UU * 4);
  bf16_t* Sbf    = (bf16_t*)alloc((size_t)NN * NN * 2);
  bf16_t* Wg0bf  = (bf16_t*)alloc((size_t)KG * 2 * UU * 2);   // transposed (128 x 384)
  bf16_t* Wc0bf  = (bf16_t*)alloc((size_t)KG * UU * 2);       // transposed (64 x 384)
  bf16_t* Wg1bf  = (bf16_t*)alloc((size_t)KG * 2 * UU * 2);
  bf16_t* Wc1bf  = (bf16_t*)alloc((size_t)KG * UU * 2);
  bf16_t* X0t    = (bf16_t*)alloc((size_t)NN * CC * 2);       // (CC x NN)
  bf16_t* X1t    = (bf16_t*)alloc((size_t)NN * CC * 2);
  bf16_t* X2t    = (bf16_t*)alloc((size_t)NN * CC * 2);
  bf16_t* Xgbf   = (bf16_t*)alloc((size_t)ROWS * KG * 2);     // (ROWS x KG)
  float*  gates  = (float*)alloc((size_t)ROWS * 2 * UU * 4);

  hipMemsetAsync(state0, 0, (size_t)ROWS * UU * 4, stream);
  hipMemsetAsync(state1, 0, (size_t)ROWS * UU * 4, stream);

  f32_to_bf16<<<(NN * NN + 255) / 256, 256, 0, stream>>>(Sbf, supports, NN * NN);
  f32_to_bf16_T<<<(KG * 2 * UU + 255) / 256, 256, 0, stream>>>(Wg0bf, Wg0, 2 * UU);
  f32_to_bf16_T<<<(KG * UU + 255) / 256, 256, 0, stream>>>(Wc0bf, Wc0, UU);
  f32_to_bf16_T<<<(KG * 2 * UU + 255) / 256, 256, 0, stream>>>(Wg1bf, Wg1, 2 * UU);
  f32_to_bf16_T<<<(KG * UU + 255) / 256, 256, 0, stream>>>(Wc1bf, Wc1, UU);

  const dim3 chebGrid(CC / 64, NN / 64);            // (64, 4) blocks of 4 waves
  const dim3 packX0Grid((NN * CC + 255) / 256);
  const dim3 packXgGrid((ROWS * KG + 255) / 256);

  auto cell = [&](const float* inp, long inp_bstride, float* state,
                  const bf16_t* Wgbf, const float* bg,
                  const bf16_t* Wcbf, const float* bc) {
    // gates pass: sigmoid(gconv([inp, state]) Wg + bg)
    pack_x0<<<packX0Grid, 256, 0, stream>>>(X0t, inp, inp_bstride, state,
                                            (const float*)nullptr);
    cheb_wmma<1, false><<<chebGrid, 128, 0, stream>>>(Sbf, X0t, X0t, X1t);
    cheb_wmma<2, true><<<chebGrid, 128, 0, stream>>>(Sbf, X1t, X0t, X2t);
    pack_xg<<<packXgGrid, 256, 0, stream>>>(Xgbf, X0t, X1t, X2t);
    proj_wmma<0><<<dim3(2, ROWS / 64), 128, 0, stream>>>(Xgbf, Wgbf, bg, gates, state);
    // candidate pass: tanh(gconv([inp, r*state]) Wc + bc) + fused GRU update
    pack_x0<<<packX0Grid, 256, 0, stream>>>(X0t, inp, inp_bstride, state, gates);
    cheb_wmma<1, false><<<chebGrid, 128, 0, stream>>>(Sbf, X0t, X0t, X1t);
    cheb_wmma<2, true><<<chebGrid, 128, 0, stream>>>(Sbf, X1t, X0t, X2t);
    pack_xg<<<packXgGrid, 256, 0, stream>>>(Xgbf, X0t, X1t, X2t);
    proj_wmma<1><<<dim3(1, ROWS / 64), 128, 0, stream>>>(Xgbf, Wcbf, bc, gates, state);
  };

  for (int t = 0; t < TT; ++t) {
    // layer 0 consumes the raw input at time t (batch stride = T*N*FIN)
    cell(input_seq + (size_t)t * NN * FIN, (long)TT * NN * FIN, state0,
         Wg0bf, bg0, Wc0bf, bc0);
    // layer 1 consumes layer 0's freshly updated state (batch stride = N*UU)
    cell(state0, (long)NN * UU, state1, Wg1bf, bg1, Wc1bf, bc1);
  }

  head_max<<<NB, 256, 0, stream>>>(state1, Wfc, bfc, (float*)d_out);
}
